// CausalSelfAttention_87282325389437
// MI455X (gfx1250) — compile-verified, matrix operand order checked by reading, not run on hardware
//
#include <hip/hip_runtime.h>
#include <hip/hip_bf16.h>

typedef __attribute__((ext_vector_type(16))) _Float16 v16h;
typedef __attribute__((ext_vector_type(8)))  float    v8f;

#define B_  4
#define T_  2048
#define C_  2048
#define H_  16
#define HD_ 128
#define F_  (3 * C_)

__device__ __forceinline__ v8f wmma16x16x32(v16h a, v16h b, v8f c) {
  // D = A(16x32 f16) * B(32x16 f16) + C(16x16 f32)
  return __builtin_amdgcn_wmma_f32_16x16x32_f16(
      /*neg_a=*/false, a, /*neg_b=*/false, b,
      /*c_mod=*/(short)0, c, /*reuse_a=*/false, /*reuse_b=*/false);
}

// ---------------------------------------------------------------------------
// Elementwise f32 -> f16 conversion (feeds the matrix cores)
// ---------------------------------------------------------------------------
__global__ void cvt_f32_to_f16(const float* __restrict__ in,
                               _Float16* __restrict__ out, long n) {
  long i = (long)blockIdx.x * blockDim.x + threadIdx.x;
  long stride = (long)gridDim.x * blockDim.x;
  for (; i < n; i += stride) out[i] = (_Float16)in[i];
}

// ---------------------------------------------------------------------------
// NT GEMM: C[M,N] = A[M,K] * B[N,K]^T   (both row-major along K, f16 in,
// f32 accumulate via WMMA, OutT output). 256 threads = 8 waves.
// Tile 128(M) x 128(N), K-step 32. Each wave computes 32x64 (2x4 frags).
// ---------------------------------------------------------------------------
template <typename OutT>
__global__ __launch_bounds__(256) void gemm_nt_wmma(
    const _Float16* __restrict__ A, const _Float16* __restrict__ Bm,
    OutT* __restrict__ Cmat, int M, int N, int K) {
  constexpr int BM = 128, BN = 128, BK = 32, PAD = 16;  // stride 48 f16 = 96B
  __shared__ _Float16 As[BM][BK + PAD];
  __shared__ _Float16 Bs[BN][BK + PAD];

  const int mb = blockIdx.y * BM;
  const int nb = blockIdx.x * BN;
  const int tid = threadIdx.x;
  const int wid = tid >> 5;
  const int lane = tid & 31;
  const int wm = (wid & 3) * 32;   // 4 waves along M
  const int wn = (wid >> 2) * 64;  // 2 waves along N
  const int lrow = lane & 15;
  const int lko = (lane >> 4) * 16;
  const int hi8 = (lane >> 4) << 3;

  v8f acc[2][4] = {};

  const int r = tid >> 1;          // 128 rows, 2 threads per row
  const int hh = (tid & 1) * 16;   // each thread: 16 f16 = 32B

  for (int k0 = 0; k0 < K; k0 += BK) {
    const _Float16* ga = A + (size_t)(mb + r) * K + k0 + hh;
    const _Float16* gb = Bm + (size_t)(nb + r) * K + k0 + hh;
    *(float4*)&As[r][hh] = *(const float4*)ga;
    *(float4*)&As[r][hh + 8] = *(const float4*)(ga + 8);
    *(float4*)&Bs[r][hh] = *(const float4*)gb;
    *(float4*)&Bs[r][hh + 8] = *(const float4*)(gb + 8);
    if (k0 + BK < K) {  // prefetch next K tile (global_prefetch_b8)
      __builtin_prefetch(ga + BK, 0, 1);
      __builtin_prefetch(gb + BK, 0, 1);
    }
    __syncthreads();

    v16h af[2], bf[4];
#pragma unroll
    for (int i = 0; i < 2; i++)
      af[i] = *(const v16h*)&As[wm + i * 16 + lrow][lko];
#pragma unroll
    for (int j = 0; j < 4; j++)
      bf[j] = *(const v16h*)&Bs[wn + j * 16 + lrow][lko];
#pragma unroll
    for (int i = 0; i < 2; i++)
#pragma unroll
      for (int j = 0; j < 4; j++)
        acc[i][j] = wmma16x16x32(af[i], bf[j], acc[i][j]);
    __syncthreads();
  }

  // C/D layout: VGPR e, lanes 0-15 -> row e, lanes 16-31 -> row e+8; col=lane%16
#pragma unroll
  for (int i = 0; i < 2; i++)
#pragma unroll
    for (int j = 0; j < 4; j++)
#pragma unroll
      for (int e = 0; e < 8; e++) {
        int row = mb + wm + i * 16 + e + hi8;
        int col = nb + wn + j * 16 + lrow;
        Cmat[(size_t)row * N + col] = (OutT)acc[i][j][e];
      }
}

// ---------------------------------------------------------------------------
// RoPE + head split. Reads qkv f16 [B,T,3C]; writes Q (scaled by 1/sqrt(hd))
// and K as [B,H,T,HD] f16; writes V transposed [B,H,HD,T] f16 so that the
// PV WMMA B-fragments are contiguous loads.
// ---------------------------------------------------------------------------
__global__ __launch_bounds__(256) void rope_split(
    const _Float16* __restrict__ qkv, _Float16* __restrict__ Q,
    _Float16* __restrict__ Kd, _Float16* __restrict__ Vt) {
  long idx = (long)blockIdx.x * blockDim.x + threadIdx.x;
  if (idx >= (long)B_ * T_ * H_ * (HD_ / 2)) return;
  int i = (int)(idx & 63);            // rotation pair index
  int h = (int)((idx >> 6) & (H_ - 1));
  int t = (int)((idx >> 10) & (T_ - 1));
  int b = (int)(idx >> 21);

  const _Float16* base = qkv + (size_t)(b * T_ + t) * F_;
  int c = h * HD_ + 2 * i;
  float q0 = (float)base[c], q1 = (float)base[c + 1];
  float k0 = (float)base[C_ + c], k1 = (float)base[C_ + c + 1];
  float v0 = (float)base[2 * C_ + c], v1 = (float)base[2 * C_ + c + 1];

  // theta = 10000^(-2i/HD) = exp(-2i/HD * ln 10000)
  float theta = __expf(-(float)(2 * i) * (9.210340371976184f / (float)HD_));
  float sn, cs;
  __sincosf((float)t * theta, &sn, &cs);
  const float scale = 0.08838834764831845f;  // 1/sqrt(128), folded into Q

  size_t headbase = ((size_t)(b * H_ + h) * T_ + t) * HD_;
  Q[headbase + 2 * i]     = (_Float16)((q0 * cs - q1 * sn) * scale);
  Q[headbase + 2 * i + 1] = (_Float16)((q0 * sn + q1 * cs) * scale);
  Kd[headbase + 2 * i]     = (_Float16)(k0 * cs - k1 * sn);
  Kd[headbase + 2 * i + 1] = (_Float16)(k0 * sn + k1 * cs);

  size_t vtbase = ((size_t)(b * H_ + h) * HD_ + 2 * i) * T_ + t;
  Vt[vtbase] = (_Float16)v0;
  Vt[vtbase + T_] = (_Float16)v1;
}

// ---------------------------------------------------------------------------
// Flash attention (causal, online softmax). One block per (b, h, 128-query
// tile); 8 waves, each owns 16 query rows. K/V staged in LDS per 32-key
// chunk; QK^T and PV on WMMA; softmax stats in f32 VALU with shfl_xor row
// reductions (N dim is striped across the 16-lane halves of the wave).
// ---------------------------------------------------------------------------
__global__ __launch_bounds__(256) void flash_attn(
    const _Float16* __restrict__ Q, const _Float16* __restrict__ Kd,
    const _Float16* __restrict__ Vt, _Float16* __restrict__ Y) {
  constexpr int PAD = 16;
  __shared__ _Float16 Ks[32][HD_ + PAD];       // 32 keys x 128 hd
  __shared__ _Float16 Vs[HD_][32 + PAD];       // 128 hd x 32 keys (transposed V)
  __shared__ _Float16 Ps[8][16][32 + PAD];     // per-wave P staging (16q x 32k)

  const int b = blockIdx.z, h = blockIdx.y;
  const int q0 = blockIdx.x * 128;
  const int tid = threadIdx.x, wid = tid >> 5, lane = tid & 31;
  const int qbase = q0 + wid * 16;
  const int lrow = lane & 15;
  const int lko = (lane >> 4) * 16;
  const int hi8 = (lane >> 4) << 3;

  const size_t headQK = (size_t)(b * H_ + h) * T_ * HD_;
  const size_t headV = (size_t)(b * H_ + h) * HD_ * T_;

  // Q fragments for this wave's 16 rows stay in registers (4 k-chunks of 32)
  v16h qf[4];
#pragma unroll
  for (int c = 0; c < 4; c++)
    qf[c] = *(const v16h*)(Q + headQK + (size_t)(qbase + lrow) * HD_ + c * 32 + lko);

  v8f o[8] = {};          // 16q x 128hd accumulator (8 x 16-col tiles)
  float mrow[8], lsum[8];
#pragma unroll
  for (int e = 0; e < 8; e++) { mrow[e] = -3.0e38f; lsum[e] = 0.f; }

  for (int kb = 0; kb < q0 + 128; kb += 32) {
    {  // cooperative K / Vt tile loads
      int kr = tid >> 3, seg = (tid & 7) * 16;
      const _Float16* gk = Kd + headQK + (size_t)(kb + kr) * HD_ + seg;
      *(float4*)&Ks[kr][seg] = *(const float4*)gk;
      *(float4*)&Ks[kr][seg + 8] = *(const float4*)(gk + 8);
      int vr = tid >> 1, vseg = (tid & 1) * 16;
      const _Float16* gv = Vt + headV + (size_t)vr * T_ + kb + vseg;
      *(float4*)&Vs[vr][vseg] = *(const float4*)gv;
      *(float4*)&Vs[vr][vseg + 8] = *(const float4*)(gv + 8);
    }
    __syncthreads();

    if (kb < qbase + 16) {  // causally relevant for this wave's rows
      v8f s[2] = {};
#pragma unroll
      for (int c = 0; c < 4; c++) {
#pragma unroll
        for (int t2 = 0; t2 < 2; t2++) {
          v16h kf = *(const v16h*)&Ks[t2 * 16 + lrow][c * 32 + lko];
          s[t2] = wmma16x16x32(qf[c], kf, s[t2]);
        }
      }
#pragma unroll
      for (int e = 0; e < 8; e++) {
        int qrow = qbase + e + hi8;
#pragma unroll
        for (int t2 = 0; t2 < 2; t2++) {
          int key = kb + t2 * 16 + lrow;
          if (key > qrow) s[t2][e] = -3.0e38f;  // causal mask
        }
        float mx = fmaxf(s[0][e], s[1][e]);
#pragma unroll
        for (int off = 1; off < 16; off <<= 1)
          mx = fmaxf(mx, __shfl_xor(mx, off, 32));
        mx = fmaxf(mx, mrow[e]);
        float alpha = __expf(mrow[e] - mx);
        mrow[e] = mx;
        float p0 = __expf(s[0][e] - mx);
        float p1 = __expf(s[1][e] - mx);
        float ps = p0 + p1;
#pragma unroll
        for (int off = 1; off < 16; off <<= 1)
          ps += __shfl_xor(ps, off, 32);
        lsum[e] = lsum[e] * alpha + ps;
#pragma unroll
        for (int j = 0; j < 8; j++) o[j][e] *= alpha;
        Ps[wid][e + hi8][lrow] = (_Float16)p0;
        Ps[wid][e + hi8][16 + lrow] = (_Float16)p1;
      }
      // restructure P into an A-fragment via per-wave LDS (in-order w/ stores)
      v16h pf = *(const v16h*)&Ps[wid][lrow][lko];
#pragma unroll
      for (int j = 0; j < 8; j++) {
        v16h vf = *(const v16h*)&Vs[j * 16 + lrow][lko];
        o[j] = wmma16x16x32(pf, vf, o[j]);
      }
    }
    __syncthreads();
  }

  // normalize and write Y[B,T,C] f16 (head-major columns)
#pragma unroll
  for (int e = 0; e < 8; e++) {
    float inv = 1.0f / lsum[e];
    int t = qbase + e + hi8;
#pragma unroll
    for (int j = 0; j < 8; j++)
      Y[(size_t)(b * T_ + t) * C_ + h * HD_ + j * 16 + lrow] =
          (_Float16)(o[j][e] * inv);
  }
}

// ---------------------------------------------------------------------------
extern "C" void kernel_launch(void* const* d_in, const int* in_sizes, int n_in,
                              void* d_out, int out_size, void* d_ws,
                              size_t ws_size, hipStream_t stream) {
  const float* x = (const float*)d_in[0];
  const float* w_attn = (const float*)d_in[1];
  const float* w_proj = (const float*)d_in[2];
  float* out = (float*)d_out;

  char* ws = (char*)d_ws;
  size_t off = 0;
  auto take = [&](size_t bytes) -> char* {
    char* p = ws + off;
    off = (off + bytes + 255) & ~(size_t)255;
    return p;
  };
  _Float16* Xh = (_Float16*)take((size_t)B_ * T_ * C_ * 2);
  _Float16* Wah = (_Float16*)take((size_t)F_ * C_ * 2);
  _Float16* Wph = (_Float16*)take((size_t)C_ * C_ * 2);
  _Float16* QKVh = (_Float16*)take((size_t)B_ * T_ * F_ * 2);
  _Float16* Q = (_Float16*)take((size_t)B_ * H_ * T_ * HD_ * 2);
  _Float16* Kd = (_Float16*)take((size_t)B_ * H_ * T_ * HD_ * 2);
  _Float16* Vt = (_Float16*)take((size_t)B_ * H_ * T_ * HD_ * 2);
  _Float16* Y = QKVh;  // QKV is fully consumed by rope_split before Y written

  cvt_f32_to_f16<<<4096, 256, 0, stream>>>(x, Xh, (long)B_ * T_ * C_);
  cvt_f32_to_f16<<<4096, 256, 0, stream>>>(w_attn, Wah, (long)F_ * C_);
  cvt_f32_to_f16<<<4096, 256, 0, stream>>>(w_proj, Wph, (long)C_ * C_);

  // QKV projection: [8192,2048] x [6144,2048]^T -> [8192,6144] f16
  dim3 g1(F_ / 128, (B_ * T_) / 128);
  gemm_nt_wmma<_Float16><<<g1, 256, 0, stream>>>(Xh, Wah, QKVh, B_ * T_, F_, C_);

  long nrope = (long)B_ * T_ * H_ * (HD_ / 2);
  rope_split<<<(int)(nrope / 256), 256, 0, stream>>>(QKVh, Q, Kd, Vt);

  dim3 g2(T_ / 128, H_, B_);
  flash_attn<<<g2, 256, 0, stream>>>(Q, Kd, Vt, Y);

  // Output projection: [8192,2048] x [2048,2048]^T -> f32 d_out
  dim3 g3(C_ / 128, (B_ * T_) / 128);
  gemm_nt_wmma<float><<<g3, 256, 0, stream>>>(Y, Wph, out, B_ * T_, C_, C_);

  (void)in_sizes; (void)n_in; (void)out_size; (void)ws_size;
}
